// ConvBlock_13889924235401
// MI455X (gfx1250) — compile-verified, matrix operand order checked by reading, not run on hardware
//
#include <hip/hip_runtime.h>

typedef __attribute__((ext_vector_type(16))) _Float16 v16h;
typedef __attribute__((ext_vector_type(8)))  _Float16 v8h;
typedef __attribute__((ext_vector_type(8)))  float    v8f;

#define C_CH   64
#define HH     256
#define WW     256
#define HW     65536          // 256*256
#define CHW    4194304        // 64*HW
#define TSZ    8388608        // one coset tensor, B*C*H*W
#define NGRP   16             // B * GROUPS
#define GROUP_N 1048576       // 8ch * 2 cosets * HW
#define EPSV   1e-5f

// ---------------- weight prep: build per-lane WMMA A-fragments ----------------
// A-frag layout (16-bit A 16x32, wave32): lane L: M = L%16,
// half h: K = ks*32 + (h<8 ? h : h+8) + (L>=16 ? 8 : 0)
// K mapping: K<64 -> center weight w_c[o][K]; else tap t=(K-64)>>6, i=(K-64)&63,
// ky=t>>1, kx=t&1 -> w_k[o][i][ky][kx]
__device__ __forceinline__ float qc_weight(int K, int M, const float* wc, const float* wk) {
    if (K < 64) return wc[M * 64 + K];
    int kk = K - 64;
    int tap = kk >> 6;
    int i = kk & 63;
    return wk[((M * 64 + i) * 2 + (tap >> 1)) * 2 + (tap & 1)];
}

__global__ void prep_weights_kernel(const float* __restrict__ w1c, const float* __restrict__ w1k,
                                    const float* __restrict__ w2c, const float* __restrict__ w2k,
                                    const float* __restrict__ wsk,
                                    _Float16* __restrict__ A1, _Float16* __restrict__ A2,
                                    _Float16* __restrict__ AS) {
    // conv layers: 4 mt * 10 ks * 32 lanes * 16 halves = 20480
    for (int idx = threadIdx.x; idx < 20480; idx += 256) {
        int h    = idx & 15;
        int lane = (idx >> 4) & 31;
        int rest = idx >> 9;           // mt*10 + ks
        int ks   = rest % 10;
        int mt   = rest / 10;
        int M = mt * 16 + (lane & 15);
        int K = ks * 32 + ((h < 8) ? h : h + 8) + ((lane & 16) ? 8 : 0);
        A1[idx] = (_Float16)qc_weight(K, M, w1c, w1k);
        A2[idx] = (_Float16)qc_weight(K, M, w2c, w2k);
    }
    // skip (1x1): 4 mt * 2 ks * 32 * 16 = 4096
    for (int idx = threadIdx.x; idx < 4096; idx += 256) {
        int h    = idx & 15;
        int lane = (idx >> 4) & 31;
        int ks   = (idx >> 9) & 1;
        int mt   = idx >> 10;
        int M = mt * 16 + (lane & 15);
        int K = ks * 32 + ((h < 8) ? h : h + 8) + ((lane & 16) ? 8 : 0);
        AS[idx] = (_Float16)wsk[M * 64 + K];
    }
}

// ---------------- groupnorm stats: two-stage deterministic reduction ----------------
// grid: (32 chunks, 16 (b,g)); channels of a group are contiguous (524288 floats / tensor)
__global__ void stats_part_kernel(const float* __restrict__ t0, const float* __restrict__ t1,
                                  float* __restrict__ part) {
    int chunk = blockIdx.x, bg = blockIdx.y;
    int b = bg >> 3, g = bg & 7;
    int base = b * CHW + g * 524288 + chunk * 16384;
    float s = 0.f, q = 0.f;
    for (int i = threadIdx.x; i < 16384; i += 256) {
        float v = t0[base + i]; s += v; q += v * v;
        v = t1[base + i];       s += v; q += v * v;
    }
    __shared__ float rs[256], rq[256];
    rs[threadIdx.x] = s; rq[threadIdx.x] = q;
    __syncthreads();
    for (int off = 128; off > 0; off >>= 1) {
        if (threadIdx.x < off) { rs[threadIdx.x] += rs[threadIdx.x + off]; rq[threadIdx.x] += rq[threadIdx.x + off]; }
        __syncthreads();
    }
    if (threadIdx.x == 0) {
        part[(bg * 32 + chunk) * 2]     = rs[0];
        part[(bg * 32 + chunk) * 2 + 1] = rq[0];
    }
}

__global__ void stats_final_kernel(const float* __restrict__ part, float* __restrict__ statsOut) {
    int t = threadIdx.x;
    if (t < NGRP) {
        float s = 0.f, q = 0.f;
        for (int i = 0; i < 32; ++i) { s += part[(t * 32 + i) * 2]; q += part[(t * 32 + i) * 2 + 1]; }
        const float invN = 1.f / (float)GROUP_N;
        float mean = s * invN;
        float var  = q * invN - mean * mean;
        statsOut[t * 2]     = mean;
        statsOut[t * 2 + 1] = rsqrtf(var + EPSV);
    }
}

// ---------------- skip-path stats: s = W_skip x + b computed on the fly ----------------
__global__ void skip_stats_part_kernel(const float* __restrict__ x0, const float* __restrict__ x1,
                                       const float* __restrict__ wsk, const float* __restrict__ bsk,
                                       float* __restrict__ part) {
    int chunk = blockIdx.x, bg = blockIdx.y;
    int b = bg >> 3, g = bg & 7;
    __shared__ float wsm[8][64];
    __shared__ float bs[8];
    for (int e = threadIdx.x; e < 512; e += 256) { int o = e >> 6, i = e & 63; wsm[o][i] = wsk[(g * 8 + o) * 64 + i]; }
    if (threadIdx.x < 8) bs[threadIdx.x] = bsk[g * 8 + threadIdx.x];
    __syncthreads();
    float s = 0.f, q = 0.f;
    for (int j = 0; j < 16; ++j) {
        int slot = chunk * 4096 + j * 256 + threadIdx.x;   // 0..131071 = [coset][pixel]
        int cs = slot >> 16, px = slot & 65535;
        const float* xp = (cs ? x1 : x0) + b * CHW + px;
        float acc[8];
        #pragma unroll
        for (int o = 0; o < 8; ++o) acc[o] = bs[o];
        for (int i = 0; i < 64; ++i) {
            float xi = xp[i << 16];
            #pragma unroll
            for (int o = 0; o < 8; ++o) acc[o] += wsm[o][i] * xi;
        }
        #pragma unroll
        for (int o = 0; o < 8; ++o) { s += acc[o]; q += acc[o] * acc[o]; }
    }
    __shared__ float rs[256], rq[256];
    rs[threadIdx.x] = s; rq[threadIdx.x] = q;
    __syncthreads();
    for (int off = 128; off > 0; off >>= 1) {
        if (threadIdx.x < off) { rs[threadIdx.x] += rs[threadIdx.x + off]; rq[threadIdx.x] += rq[threadIdx.x + off]; }
        __syncthreads();
    }
    if (threadIdx.x == 0) {
        part[(bg * 32 + chunk) * 2]     = rs[0];
        part[(bg * 32 + chunk) * 2 + 1] = rq[0];
    }
}

// ---------------- fused gn+relu + quincunx conv (+ fused skip on FINAL) ----------------
// block = 256 threads (8 wave32). One output row (y) x 64 px for one (batch, coset).
// M=64 (4 tiles) x N=64 px (4 tiles) x K=320 (10 x K32 wmma).
// Wave w owns mt = w/2 and nt = (w%2)*2 + {0,1}: all 10 A-fragments are
// register-resident across both N-tiles -> steady-state loop = 2x ds_read_b128 + wmma.
template<bool FINAL>
__global__ void qc_conv_kernel(const float* __restrict__ in0, const float* __restrict__ in1,
                               const float* __restrict__ stats,
                               const float* __restrict__ gamma, const float* __restrict__ beta,
                               const _Float16* __restrict__ afrag, const float* __restrict__ bias,
                               float* __restrict__ out0, float* __restrict__ out1,
                               const float* __restrict__ rx0, const float* __restrict__ rx1,
                               const _Float16* __restrict__ afragS, const float* __restrict__ biasS,
                               const float* __restrict__ statsS,
                               const float* __restrict__ gS, const float* __restrict__ betaS) {
    __shared__ _Float16 sSame[64][64]  __attribute__((aligned(16)));    // [px][ch]
    __shared__ _Float16 sOther[2][66][64] __attribute__((aligned(16))); // [row][px0..64][ch]
    __shared__ _Float16 sRaw[64][64]   __attribute__((aligned(16)));    // raw same-coset (skip)

    const int x0 = blockIdx.x * 64;
    const int y  = blockIdx.y;
    const int b  = blockIdx.z >> 1;
    const int cs = blockIdx.z & 1;
    const int dy0 = cs ? 0 : -1;
    const int dx0 = cs ? 0 : -1;

    const float* same  = cs ? in1 : in0;
    const float* other = cs ? in0 : in1;
    float* outPtr = cs ? out1 : out0;

    // ---- stage same-coset row: gn + relu -> f16 ----
    {
        int c   = threadIdx.x >> 2;
        int pxb = (threadIdx.x & 3) << 4;
        int g = c >> 3;
        float mean = stats[(b * 8 + g) * 2];
        float inv  = stats[(b * 8 + g) * 2 + 1];
        float sc = inv * gamma[c];
        float sh = beta[c] - mean * sc;
        const float* p = same + ((b * 64 + c) * 256 + y) * 256 + x0 + pxb;
        #pragma unroll
        for (int j4 = 0; j4 < 4; ++j4) {
            float4 v = *(const float4*)(p + j4 * 4);
            sSame[pxb + j4 * 4 + 0][c] = (_Float16)fmaxf(v.x * sc + sh, 0.f);
            sSame[pxb + j4 * 4 + 1][c] = (_Float16)fmaxf(v.y * sc + sh, 0.f);
            sSame[pxb + j4 * 4 + 2][c] = (_Float16)fmaxf(v.z * sc + sh, 0.f);
            sSame[pxb + j4 * 4 + 3][c] = (_Float16)fmaxf(v.w * sc + sh, 0.f);
        }
        if constexpr (FINAL) {
            const float* rp = (cs ? rx1 : rx0) + ((b * 64 + c) * 256 + y) * 256 + x0 + pxb;
            #pragma unroll
            for (int j4 = 0; j4 < 4; ++j4) {
                float4 v = *(const float4*)(rp + j4 * 4);
                sRaw[pxb + j4 * 4 + 0][c] = (_Float16)v.x;
                sRaw[pxb + j4 * 4 + 1][c] = (_Float16)v.y;
                sRaw[pxb + j4 * 4 + 2][c] = (_Float16)v.z;
                sRaw[pxb + j4 * 4 + 3][c] = (_Float16)v.w;
            }
        }
    }
    // ---- stage other-coset 2x65 window, zero-filled at borders (lattice padding) ----
    for (int e = threadIdx.x; e < 2 * 64 * 65; e += 256) {
        int r   = e / (64 * 65);
        int rem = e % (64 * 65);
        int c   = rem / 65;
        int po  = rem % 65;
        int gy = y + dy0 + r;
        int gx = x0 + dx0 + po;
        float val = 0.f;
        if (gy >= 0 && gy < HH && gx >= 0 && gx < WW) {
            float v = other[((b * 64 + c) * 256 + gy) * 256 + gx];
            int g = c >> 3;
            float mean = stats[(b * 8 + g) * 2];
            float inv  = stats[(b * 8 + g) * 2 + 1];
            val = fmaxf((v - mean) * inv * gamma[c] + beta[c], 0.f);
        }
        sOther[r][po][c] = (_Float16)val;
    }
    __syncthreads();

    // ---- WMMA GEMM ----
    const int lane = threadIdx.x & 31;
    const int wv   = threadIdx.x >> 5;
    const int pl   = lane & 15;
    const int hi8  = (lane & 16) ? 8 : 0;
    const int mt     = wv >> 1;           // fixed output-channel tile per wave
    const int ntBase = (wv & 1) << 1;     // two adjacent pixel tiles per wave

    // weights: register-resident A fragments, loaded once
    v16h aF[10];
    #pragma unroll
    for (int ks = 0; ks < 10; ++ks)
        aF[ks] = *(const v16h*)(afrag + (((mt * 10 + ks) * 32 + lane) << 4));
    v16h aFS[2];
    if constexpr (FINAL) {
        #pragma unroll
        for (int ks = 0; ks < 2; ++ks)
            aFS[ks] = *(const v16h*)(afragS + (((mt * 2 + ks) * 32 + lane) << 4));
    }

    #pragma unroll
    for (int nti = 0; nti < 2; ++nti) {
        const int nt = ntBase + nti;
        v8f acc = {};
        v8f accS = {};
        #pragma unroll
        for (int ks = 0; ks < 10; ++ks) {
            const _Float16* bsrc;
            if (ks < 2) {
                bsrc = &sSame[nt * 16 + pl][ks * 32 + hi8];
            } else {
                int tp = (ks - 2) >> 1;          // tap 0..3 -> (ky=tp>>1, kx=tp&1)
                int kh = (ks - 2) & 1;           // K-half of the tap's 64 channels
                bsrc = &sOther[tp >> 1][nt * 16 + pl + (tp & 1)][kh * 32 + hi8];
            }
            v8h blo = *(const v8h*)bsrc;
            v8h bhi = *(const v8h*)(bsrc + 16);
            v16h bf = __builtin_shufflevector(blo, bhi, 0,1,2,3,4,5,6,7,8,9,10,11,12,13,14,15);
            acc = __builtin_amdgcn_wmma_f32_16x16x32_f16(false, aF[ks], false, bf, (short)0, acc, false, false);
        }
        if constexpr (FINAL) {
            #pragma unroll
            for (int ks = 0; ks < 2; ++ks) {
                const _Float16* bsrc = &sRaw[nt * 16 + pl][ks * 32 + hi8];
                v8h blo = *(const v8h*)bsrc;
                v8h bhi = *(const v8h*)(bsrc + 16);
                v16h bf = __builtin_shufflevector(blo, bhi, 0,1,2,3,4,5,6,7,8,9,10,11,12,13,14,15);
                accS = __builtin_amdgcn_wmma_f32_16x16x32_f16(false, aFS[ks], false, bf, (short)0, accS, false, false);
            }
        }
        const int px = nt * 16 + pl;
        #pragma unroll
        for (int r = 0; r < 8; ++r) {
            int o = mt * 16 + r + hi8;            // D layout: VGPR r -> M = r + (lane>=16 ? 8:0)
            float val = acc[r] + bias[o];
            if constexpr (FINAL) {
                float sv = accS[r] + biasS[o];
                int gg = o >> 3;
                float mS = statsS[(b * 8 + gg) * 2];
                float iS = statsS[(b * 8 + gg) * 2 + 1];
                val += (sv - mS) * iS * gS[o] + betaS[o];
            }
            outPtr[((b * 64 + o) * 256 + y) * 256 + x0 + px] = val;
        }
    }
}

// ---------------- launcher ----------------
extern "C" void kernel_launch(void* const* d_in, const int* in_sizes, int n_in,
                              void* d_out, int out_size, void* d_ws, size_t ws_size,
                              hipStream_t stream) {
    const float* x0   = (const float*)d_in[0];
    const float* x1   = (const float*)d_in[1];
    const float* g1   = (const float*)d_in[2];
    const float* b1   = (const float*)d_in[3];
    const float* w1c  = (const float*)d_in[4];
    const float* w1k  = (const float*)d_in[5];
    const float* bias1= (const float*)d_in[6];
    const float* g2   = (const float*)d_in[7];
    const float* b2   = (const float*)d_in[8];
    const float* w2c  = (const float*)d_in[9];
    const float* w2k  = (const float*)d_in[10];
    const float* bias2= (const float*)d_in[11];
    const float* wsk  = (const float*)d_in[12];
    const float* bsk  = (const float*)d_in[13];
    const float* gsk  = (const float*)d_in[14];
    const float* btk  = (const float*)d_in[15];

    char* ws = (char*)d_ws;
    float*    wsH    = (float*)(ws);                        // h0,h1: 2*TSZ f32 = 64 MiB
    _Float16* A1     = (_Float16*)(ws + 67108864);          // 40960 B
    _Float16* A2     = (_Float16*)(ws + 67108864 + 40960);  // 40960 B
    _Float16* AS     = (_Float16*)(ws + 67108864 + 81920);  // 8192 B
    float*    stats1 = (float*)(ws + 67108864 + 90112);     // 128 B each
    float*    stats2 = stats1 + 32;
    float*    statsS = stats2 + 32;
    float*    part   = statsS + 32;                         // 1024 f32, reused per phase

    float* outF = (float*)d_out;

    prep_weights_kernel<<<1, 256, 0, stream>>>(w1c, w1k, w2c, w2k, wsk, A1, A2, AS);

    // stage 1: stats(x) -> gn+relu+conv1 -> h (ws)
    stats_part_kernel<<<dim3(32, 16), 256, 0, stream>>>(x0, x1, part);
    stats_final_kernel<<<1, 256, 0, stream>>>(part, stats1);
    qc_conv_kernel<false><<<dim3(4, 256, 4), 256, 0, stream>>>(
        x0, x1, stats1, g1, b1, A1, bias1, wsH, wsH + TSZ,
        nullptr, nullptr, nullptr, nullptr, nullptr, nullptr, nullptr);

    // stage 2 stats + skip stats
    stats_part_kernel<<<dim3(32, 16), 256, 0, stream>>>(wsH, wsH + TSZ, part);
    stats_final_kernel<<<1, 256, 0, stream>>>(part, stats2);
    skip_stats_part_kernel<<<dim3(32, 16), 256, 0, stream>>>(x0, x1, wsk, bsk, part);
    stats_final_kernel<<<1, 256, 0, stream>>>(part, statsS);

    // final: gn+relu+conv2 fused with skip GEMM + skip-gn + residual add -> d_out
    qc_conv_kernel<true><<<dim3(4, 256, 4), 256, 0, stream>>>(
        wsH, wsH + TSZ, stats2, g2, b2, A2, bias2, outF, outF + TSZ,
        x0, x1, AS, bsk, statsS, gsk, btk);
}
